// MHA_25211458027854
// MI455X (gfx1250) — matrix operand order repaired and checked by reading, not compiled, not run on hardware
//
#include <hip/hip_runtime.h>

typedef __bf16 bf16_t;
typedef __attribute__((ext_vector_type(16))) __bf16 v16bf;
typedef __attribute__((ext_vector_type(8)))  __bf16 v8bf;
typedef __attribute__((ext_vector_type(8)))  float  v8f;
typedef int v4i_t __attribute__((vector_size(16)));

static constexpr int BB   = 2;
static constexpr int LSEQ = 2048;
static constexpr int DMOD = 2048;
static constexpr int NH   = 16;
static constexpr int DH   = 128;
#define NEG_BIG (-4294967295.0f)

#if defined(__has_builtin)
#if __has_builtin(__builtin_amdgcn_global_load_async_to_lds_b128) && \
    __has_builtin(__builtin_amdgcn_s_wait_asynccnt)
#define USE_ASYNC_LDS 1
#endif
#endif

__device__ __forceinline__ v16bf mk16(v8bf lo, v8bf hi) {
  v16bf r;
#pragma unroll
  for (int i = 0; i < 8; ++i) { r[i] = lo[i]; r[i + 8] = hi[i]; }
  return r;
}

__device__ __forceinline__ v8f wmma_bf16(v16bf a, v16bf b, v8f c) {
  return __builtin_amdgcn_wmma_f32_16x16x32_bf16(false, a, false, b, (short)0, c,
                                                 false, false);
}

// ---------------- fp32 -> bf16 ----------------
__global__ void cvt_f32_bf16(const float* __restrict__ in, bf16_t* __restrict__ out, int n) {
  int i = blockIdx.x * blockDim.x + threadIdx.x;
  if (i < n) out[i] = (bf16_t)in[i];
}

// ---------------- tiled WMMA GEMM: C = A[M,K]*B[K,N] + bias ----------------
// MODE 0: bf16 row-major. MODE 1: bf16 transposed to Vt[b,h,dh,l]. MODE 2: f32 row-major.
template <int MODE>
__global__ __launch_bounds__(256) void gemm_bf16_wmma(
    const bf16_t* __restrict__ A, const bf16_t* __restrict__ Bw,
    const float* __restrict__ bias, void* __restrict__ Cout,
    int M, int N, int Kdim) {
  const int tid  = threadIdx.x;
  const int lane = tid & 31;
  const int wid  = tid >> 5;
  const int wm   = wid & 3;   // 4 wave-rows of 32
  const int wn   = wid >> 2;  // 2 wave-cols of 64
  const int mb   = blockIdx.y * 128;
  const int nb   = blockIdx.x * 128;

  __shared__ __align__(16) bf16_t As[128][64];
  __shared__ __align__(16) bf16_t Bst[128][64];  // [n][k] transposed

  v8f acc[2][4];
#pragma unroll
  for (int i = 0; i < 2; ++i)
#pragma unroll
    for (int j = 0; j < 4; ++j) acc[i][j] = {};

  const int l16  = lane & 15;
  const int hihf = lane >> 4;        // 0|1
  const int aoff = hihf ? 16 : 0;    // A frag K chunk select
  const int bkh  = hihf ? 16 : 0;    // B frag K half select

  for (int k0 = 0; k0 < Kdim; k0 += 64) {
#ifdef USE_ASYNC_LDS
#pragma unroll
    for (int it = 0; it < 4; ++it) {  // A tile 128x64 via async DMA to LDS
      int c = it * 256 + tid;
      int r = c >> 3, kc = (c & 7) * 8;
      __builtin_amdgcn_global_load_async_to_lds_b128(
          (v4i_t*)&A[(size_t)(mb + r) * Kdim + k0 + kc],
          (v4i_t*)&As[r][kc], 0, 0);
    }
#else
#pragma unroll
    for (int it = 0; it < 4; ++it) {  // A tile 128x64
      int c = it * 256 + tid;
      int r = c >> 3, kc = (c & 7) * 8;
      *(v8bf*)&As[r][kc] = *(const v8bf*)&A[(size_t)(mb + r) * Kdim + k0 + kc];
    }
#endif
#pragma unroll
    for (int it = 0; it < 4; ++it) {  // B tile 64x128 -> transposed (needs VGPR path)
      int c = it * 256 + tid;
      int kr = c >> 4, nc = (c & 15) * 8;
      v8bf v = *(const v8bf*)&Bw[(size_t)(k0 + kr) * N + nb + nc];
#pragma unroll
      for (int j = 0; j < 8; ++j) Bst[nc + j][kr] = v[j];
    }
#ifdef USE_ASYNC_LDS
    __builtin_amdgcn_s_wait_asynccnt(0);
#endif
    __syncthreads();

#pragma unroll
    for (int ks = 0; ks < 2; ++ks) {
      const int kk = ks * 32;
      v16bf af[2];
#pragma unroll
      for (int tm = 0; tm < 2; ++tm) {
        int m = wm * 32 + tm * 16 + l16;
        af[tm] = mk16(*(const v8bf*)&As[m][kk + aoff],
                      *(const v8bf*)&As[m][kk + aoff + 8]);
      }
      v16bf bfr[4];
#pragma unroll
      for (int tn = 0; tn < 4; ++tn) {
        int n = wn * 64 + tn * 16 + l16;
        bfr[tn] = mk16(*(const v8bf*)&Bst[n][kk + bkh],
                       *(const v8bf*)&Bst[n][kk + bkh + 8]);
      }
#pragma unroll
      for (int tm = 0; tm < 2; ++tm)
#pragma unroll
        for (int tn = 0; tn < 4; ++tn)
          acc[tm][tn] = wmma_bf16(af[tm], bfr[tn], acc[tm][tn]);
    }
    __syncthreads();
  }

  const int mloc = hihf ? 8 : 0;
#pragma unroll
  for (int tm = 0; tm < 2; ++tm)
#pragma unroll
    for (int tn = 0; tn < 4; ++tn) {
      int gn = nb + wn * 64 + tn * 16 + l16;
      float bia = bias[gn];
#pragma unroll
      for (int i = 0; i < 8; ++i) {
        int gm = mb + wm * 32 + tm * 16 + mloc + i;
        float v = acc[tm][tn][i] + bia;
        if (MODE == 0) {
          ((bf16_t*)Cout)[(size_t)gm * N + gn] = (bf16_t)v;
        } else if (MODE == 1) {
          int h = gn >> 7, dh = gn & 127;
          int bidx = gm >> 11, tok = gm & 2047;
          ((bf16_t*)Cout)[((size_t)(((bidx << 4) + h) << 7) + dh) * LSEQ + tok] = (bf16_t)v;
        } else {
          ((float*)Cout)[(size_t)gm * N + gn] = v;
        }
      }
    }
}

// ---------------- flash attention (reverse-causal: keep m >= l) ----------------
// One wave32 per 16-row query tile. Q,K: bf16 [B*L, DMOD]; Vt: bf16 [B,H,DH,L]; O: bf16 [B*L, DMOD].
__global__ __launch_bounds__(128) void attn_flash_wmma(
    const bf16_t* __restrict__ Q, const bf16_t* __restrict__ K,
    const bf16_t* __restrict__ Vt, bf16_t* __restrict__ O) {
  __shared__ __align__(16) bf16_t Pl[4][16][32];
  const int lane = threadIdx.x & 31;
  const int w    = threadIdx.x >> 5;
  const int gt   = blockIdx.x * 4 + w;
  const int tile = gt & 127;  // L/16 tiles per (b,h)
  const int bh   = gt >> 7;
  const int b    = bh >> 4;
  const int h    = bh & 15;
  const int r0   = tile * 16;

  const int l16  = lane & 15;
  const int hihf = lane >> 4;
  const int aoff = hihf ? 16 : 0;
  const int bkh  = hihf ? 16 : 0;
  const int mloc = hihf ? 8 : 0;

  const bf16_t* qrow = Q + (size_t)(b * LSEQ + r0 + l16) * DMOD + h * DH;
  v16bf qf[4];
#pragma unroll
  for (int kk = 0; kk < 4; ++kk)
    qf[kk] = mk16(*(const v8bf*)(qrow + kk * 32 + aoff),
                  *(const v8bf*)(qrow + kk * 32 + aoff + 8));

  v8f o[8];
#pragma unroll
  for (int t = 0; t < 8; ++t) o[t] = {};
  float mrow[8], lrow[8];
#pragma unroll
  for (int i = 0; i < 8; ++i) { mrow[i] = -3.0e38f; lrow[i] = 0.0f; }

  const int kb0 = (r0 >> 5) << 5;  // first key block touching the diagonal
  for (int kb = kb0; kb < LSEQ; kb += 32) {
    // prefetch next key block (global_prefetch_b8)
    if (kb + 32 < LSEQ) {
      __builtin_prefetch(K + (size_t)(b * LSEQ + kb + 32 + l16) * DMOD + h * DH, 0, 0);
      __builtin_prefetch(Vt + (size_t)(bh * DH + l16 * 8) * LSEQ + kb + 32, 0, 0);
    }
    v8f s0 = {}, s1 = {};
#pragma unroll
    for (int kk = 0; kk < 4; ++kk) {
      const bf16_t* kr0 = K + (size_t)(b * LSEQ + kb + l16) * DMOD + h * DH + kk * 32 + bkh;
      const bf16_t* kr1 = kr0 + (size_t)16 * DMOD;
      v16bf b0 = mk16(*(const v8bf*)kr0, *(const v8bf*)(kr0 + 8));
      v16bf b1 = mk16(*(const v8bf*)kr1, *(const v8bf*)(kr1 + 8));
      s0 = wmma_bf16(qf[kk], b0, s0);
      s1 = wmma_bf16(qf[kk], b1, s1);
    }
    if (kb < r0 + 16) {  // partially masked diagonal block(s)
#pragma unroll
      for (int i = 0; i < 8; ++i) {
        int R = r0 + mloc + i;
        if (kb + l16 < R)      s0[i] = NEG_BIG;
        if (kb + 16 + l16 < R) s1[i] = NEG_BIG;
      }
    }
    float pscale[8];
#pragma unroll
    for (int i = 0; i < 8; ++i) {
      float rm = fmaxf(s0[i], s1[i]);
#pragma unroll
      for (int off = 1; off < 16; off <<= 1)
        rm = fmaxf(rm, __shfl_xor(rm, off, 16));
      float mnew = fmaxf(mrow[i], rm);
      float sc = __expf(mrow[i] - mnew);
      float p0 = __expf(s0[i] - mnew);
      float p1 = __expf(s1[i] - mnew);
      s0[i] = p0; s1[i] = p1;
      float rs = p0 + p1;
#pragma unroll
      for (int off = 1; off < 16; off <<= 1)
        rs += __shfl_xor(rs, off, 16);
      lrow[i] = lrow[i] * sc + rs;
      mrow[i] = mnew;
      pscale[i] = sc;
    }
#pragma unroll
    for (int t = 0; t < 8; ++t)
#pragma unroll
      for (int i = 0; i < 8; ++i) o[t][i] *= pscale[i];

    // stage P through LDS to reach the A-fragment layout
#pragma unroll
    for (int i = 0; i < 8; ++i) {
      Pl[w][mloc + i][l16]      = (bf16_t)s0[i];
      Pl[w][mloc + i][16 + l16] = (bf16_t)s1[i];
    }
    asm volatile("s_wait_dscnt 0" ::: "memory");
    v16bf pf = mk16(*(const v8bf*)&Pl[w][l16][aoff],
                    *(const v8bf*)&Pl[w][l16][aoff + 8]);
#pragma unroll
    for (int t = 0; t < 8; ++t) {
      const bf16_t* vr = Vt + (size_t)(bh * DH + t * 16 + l16) * LSEQ + kb + bkh;
      v16bf vf = mk16(*(const v8bf*)vr, *(const v8bf*)(vr + 8));
      o[t] = wmma_bf16(pf, vf, o[t]);
    }
  }

  float invl[8];
#pragma unroll
  for (int i = 0; i < 8; ++i) invl[i] = 1.0f / (lrow[i] * 11.3137084989848f);  // * sqrt(128)
#pragma unroll
  for (int t = 0; t < 8; ++t)
#pragma unroll
    for (int i = 0; i < 8; ++i) {
      int gm = r0 + mloc + i;
      O[(size_t)(b * LSEQ + gm) * DMOD + h * DH + t * 16 + l16] = (bf16_t)(o[t][i] * invl[i]);
    }
}

extern "C" void kernel_launch(void* const* d_in, const int* in_sizes, int n_in,
                              void* d_out, int out_size, void* d_ws, size_t ws_size,
                              hipStream_t stream) {
  (void)in_sizes; (void)n_in; (void)out_size; (void)ws_size;
  const float* X  = (const float*)d_in[0];
  const float* Wq = (const float*)d_in[1];
  const float* bq = (const float*)d_in[2];
  const float* Wk = (const float*)d_in[3];
  const float* bk = (const float*)d_in[4];
  const float* Wv = (const float*)d_in[5];
  const float* bv = (const float*)d_in[6];
  const float* Wo = (const float*)d_in[7];
  const float* bo = (const float*)d_in[8];

  const size_t nX = (size_t)BB * LSEQ * DMOD;  // 8,388,608
  const size_t nW = (size_t)DMOD * DMOD;       // 4,194,304
  char* ws = (char*)d_ws;
  bf16_t* Xb  = (bf16_t*)ws; ws += nX * 2;
  bf16_t* Wqb = (bf16_t*)ws; ws += nW * 2;
  bf16_t* Wkb = (bf16_t*)ws; ws += nW * 2;
  bf16_t* Wvb = (bf16_t*)ws; ws += nW * 2;
  bf16_t* Wob = (bf16_t*)ws; ws += nW * 2;
  bf16_t* Qb  = (bf16_t*)ws; ws += nX * 2;
  bf16_t* Kb  = (bf16_t*)ws; ws += nX * 2;
  bf16_t* Vtb = (bf16_t*)ws; ws += nX * 2;
  bf16_t* Ob  = (bf16_t*)ws;

  cvt_f32_bf16<<<(int)((nX + 255) / 256), 256, 0, stream>>>(X,  Xb,  (int)nX);
  cvt_f32_bf16<<<(int)((nW + 255) / 256), 256, 0, stream>>>(Wq, Wqb, (int)nW);
  cvt_f32_bf16<<<(int)((nW + 255) / 256), 256, 0, stream>>>(Wk, Wkb, (int)nW);
  cvt_f32_bf16<<<(int)((nW + 255) / 256), 256, 0, stream>>>(Wv, Wvb, (int)nW);
  cvt_f32_bf16<<<(int)((nW + 255) / 256), 256, 0, stream>>>(Wo, Wob, (int)nW);

  dim3 gg(DMOD / 128, (BB * LSEQ) / 128);  // (16, 32)
  gemm_bf16_wmma<0><<<gg, 256, 0, stream>>>(Xb, Wqb, bq, (void*)Qb,  BB * LSEQ, DMOD, DMOD);
  gemm_bf16_wmma<0><<<gg, 256, 0, stream>>>(Xb, Wkb, bk, (void*)Kb,  BB * LSEQ, DMOD, DMOD);
  gemm_bf16_wmma<1><<<gg, 256, 0, stream>>>(Xb, Wvb, bv, (void*)Vtb, BB * LSEQ, DMOD, DMOD);

  attn_flash_wmma<<<(BB * NH * (LSEQ / 16)) / 4, 128, 0, stream>>>(Qb, Kb, Vtb, Ob);

  gemm_bf16_wmma<2><<<gg, 256, 0, stream>>>(Ob, Wob, bo, d_out, BB * LSEQ, DMOD, DMOD);
}
